// CausalSelfAttention_47571057771137
// MI455X (gfx1250) — matrix operand order repaired and checked by reading, not compiled, not run on hardware
//
#include <hip/hip_runtime.h>

// Problem constants (fixed by the reference)
#define TSEQ 2048
#define NB   2
#define NH   16
#define HD   64
#define CEMB 1024
#define MROWS (NB * TSEQ)   // 4096

typedef __bf16 bf16_t;
typedef bf16_t v16bf __attribute__((ext_vector_type(16)));
typedef bf16_t v8bf  __attribute__((ext_vector_type(8)));
typedef float  v8f   __attribute__((ext_vector_type(8)));

union AFrag { v16bf v; v8bf h[2]; };

// float -> bf16 bits, round-to-nearest-even
static __device__ __forceinline__ unsigned short f2bfu(float f) {
    unsigned u = __builtin_bit_cast(unsigned, f);
    u += 0x7FFFu + ((u >> 16) & 1u);
    return (unsigned short)(u >> 16);
}

// ---------------------------------------------------------------------------
// f32 -> bf16 conversion, 4 elements/thread (b128 in, b64 out)
// ---------------------------------------------------------------------------
__global__ __launch_bounds__(256) void k_cvt4(const float4* __restrict__ in,
                                              unsigned long long* __restrict__ out,
                                              int n4) {
    int i = blockIdx.x * 256 + threadIdx.x;
    if (i < n4) {
        float4 f = in[i];
        unsigned long long r =  (unsigned long long)f2bfu(f.x)
                             | ((unsigned long long)f2bfu(f.y) << 16)
                             | ((unsigned long long)f2bfu(f.z) << 32)
                             | ((unsigned long long)f2bfu(f.w) << 48);
        out[i] = r;
    }
}

// ---------------------------------------------------------------------------
// GEMM: out[m,n] = sum_k A[m,k] * W[n,k] + bias[n]   (torch Linear: x @ W^T)
//
// 256-thread block = 8 waves computes a 128x64 tile. The shared B (weight)
// tile [64 cols x 32 k] is ASYNC-copied (GLOBAL_LOAD_ASYNC_TO_LDS_B128,
// ASYNCcnt) into a double-buffered LDS tile with padded stride (40 elems =
// 80 B: 16B-aligned for b128, bank-conflict-free reads) and reused by all 8
// waves. Each wave holds 4 C fragments (16x64) so one A fragment feeds 4
// WMMAs. Async copy for k-step i+1 overlaps WMMA on k-step i.
//
// mode 0: store bf16 [B,H,T,D]  (Q, K)
// mode 1: store bf16 [B,H,D,T]  (V transposed, contiguous PV B-frags)
// mode 2: store f32  [M,N]      (final output projection)
// ---------------------------------------------------------------------------
#define BSTRIDE 40           // padded LDS stride per column (elements)
#define NKSTEP  (CEMB / 32)  // 32 k-steps

__global__ __launch_bounds__(256) void k_gemm(const unsigned short* __restrict__ A,
                                              const unsigned short* __restrict__ W,
                                              const float* __restrict__ bias,
                                              void* __restrict__ outp,
                                              int mode) {
    __shared__ __attribute__((aligned(16))) unsigned short Bs[2][64 * BSTRIDE];

    const int K    = CEMB;
    const int tid  = threadIdx.x;
    const int lane = tid & 31;
    const int wv   = tid >> 5;                 // wave 0..7
    const int bm   = blockIdx.x >> 4;          // / (CEMB/64 = 16)
    const int bn   = blockIdx.x & 15;
    const int m0   = bm * 128 + wv * 16;
    const int n0   = bn * 64;

    const int nl  = lane & 15;
    const int kbA = (lane >> 4) * 8;           // A frag: k 0-7/16-23 vs 8-15/24-31
    const int kbB = (lane >> 4) * 16;          // unused here; B frag uses the kbA k order
    const int hb  = (lane >> 4) * 8;           // C frag row base

    // staging role of this thread: 64 cols x 4 chunks of 8 bf16 (16 B)
    const int scol = tid >> 2;                 // 0..63
    const int schk = tid & 3;                  // 0..3
    const unsigned short* sgp = W + (size_t)(n0 + scol) * K + schk * 8;

    const unsigned short* ap = A + (size_t)(m0 + nl) * K;

    v8f c[4];
#pragma unroll
    for (int j = 0; j < 4; ++j) c[j] = (v8f){0.f,0.f,0.f,0.f,0.f,0.f,0.f,0.f};

    // async-copy one 64x32 B tile into LDS buffer b for k-offset k0
    auto stage = [&](int b, int k0) {
        unsigned lds  = (unsigned)(unsigned long long)(void*)
                        &Bs[b][scol * BSTRIDE + schk * 8];
        unsigned long long ga = (unsigned long long)(const void*)(sgp + k0);
        asm volatile("global_load_async_to_lds_b128 %0, %1, off"
                     :: "v"(lds), "v"(ga) : "memory");
    };

    // one k-step of WMMA work out of LDS buffer b
    auto compute = [&](int b, int k0) {
        AFrag a;
        a.h[0] = *(const v8bf*)(ap + k0 + kbA);
        a.h[1] = *(const v8bf*)(ap + k0 + 16 + kbA);
#pragma unroll
        for (int j = 0; j < 4; ++j) {
            const unsigned short* bp = &Bs[b][(j * 16 + nl) * BSTRIDE + kbA];
            AFrag bb;
            bb.h[0] = *(const v8bf*)(bp);
            bb.h[1] = *(const v8bf*)(bp + 16);
            c[j] = __builtin_amdgcn_wmma_f32_16x16x32_bf16(
                false, a.v, false, bb.v, (short)0, c[j], false, false);
        }
    };

    int buf = 0;
    stage(0, 0);
    for (int ks = 0; ks < NKSTEP - 1; ++ks) {
        stage(buf ^ 1, (ks + 1) * 32);                       // prefetch next
        asm volatile("s_wait_asynccnt 0x1" ::: "memory");    // current tile landed
        __syncthreads();
        compute(buf, ks * 32);
        __syncthreads();                                     // readers done (WAR)
        buf ^= 1;
    }
    asm volatile("s_wait_asynccnt 0x0" ::: "memory");
    __syncthreads();
    compute(buf, (NKSTEP - 1) * 32);

    // ---- epilogue ----
    if (mode == 2) {
        float* po = (float*)outp;
#pragma unroll
        for (int j = 0; j < 4; ++j) {
            const int ncol = n0 + j * 16 + nl;
            const float bnv = bias[ncol];
#pragma unroll
            for (int r = 0; r < 8; ++r) {
                int m = m0 + hb + r;
                po[(size_t)m * CEMB + ncol] = c[j][r] + bnv;
            }
        }
    } else {
        unsigned short* po = (unsigned short*)outp;
#pragma unroll
        for (int j = 0; j < 4; ++j) {
            const int ncol = n0 + j * 16 + nl;
            const float bnv = bias[ncol];
            const int h = ncol >> 6, d = ncol & 63;
#pragma unroll
            for (int r = 0; r < 8; ++r) {
                int m  = m0 + hb + r;
                int b_ = m >> 11;              // / TSEQ
                int t  = m & (TSEQ - 1);
                size_t idx = (mode == 0)
                    ? ((size_t)((b_ * NH + h) * TSEQ + t) * HD + d)
                    : ((size_t)((b_ * NH + h) * HD + d) * TSEQ + t);
                po[idx] = f2bfu(c[j][r] + bnv);
            }
        }
    }
}

// ---------------------------------------------------------------------------
// Flash-style causal attention. One wave per 16-query tile per (batch, head).
// Online softmax over 32-key blocks; S and PV via bf16 WMMA with f32
// accumulation; P transposed C-layout -> A-layout through LDS.
// ---------------------------------------------------------------------------
__global__ __launch_bounds__(32) void k_attn(const unsigned short* __restrict__ Qb,
                                             const unsigned short* __restrict__ Kb,
                                             const unsigned short* __restrict__ Vt,
                                             unsigned short* __restrict__ Ab) {
    __shared__ __attribute__((aligned(16))) unsigned short pls[16 * 32];

    const int lane = threadIdx.x & 31;
    const int qblk = blockIdx.x & (TSEQ / 16 - 1);   // 128 query tiles
    const int bh   = blockIdx.x >> 7;
    const int q0   = qblk * 16;

    const unsigned short* Q  = Qb + (size_t)bh * TSEQ * HD;
    const unsigned short* Kp = Kb + (size_t)bh * TSEQ * HD;
    const unsigned short* Vp = Vt + (size_t)bh * HD * TSEQ;

    const int nl  = lane & 15;
    const int kbA = (lane >> 4) * 8;
    const int kbB = (lane >> 4) * 16;
    const int hb  = (lane >> 4) * 8;

    AFrag qf[2];
#pragma unroll
    for (int c = 0; c < 2; ++c) {
        const unsigned short* qp = Q + (size_t)(q0 + nl) * HD + c * 32;
        qf[c].h[0] = *(const v8bf*)(qp + kbA);
        qf[c].h[1] = *(const v8bf*)(qp + 16 + kbA);
    }

    const v8f vzero = {0.f, 0.f, 0.f, 0.f, 0.f, 0.f, 0.f, 0.f};
    v8f o[4];
#pragma unroll
    for (int i = 0; i < 4; ++i) o[i] = vzero;
    float mi[8], li[8];
#pragma unroll
    for (int r = 0; r < 8; ++r) { mi[r] = -1e30f; li[r] = 0.f; }

    for (int kk = 0; kk < q0 + 16; kk += 32) {
        v8f s[2]; s[0] = vzero; s[1] = vzero;
#pragma unroll
        for (int half = 0; half < 2; ++half) {
            int tcol = kk + half * 16 + nl;
            if (tcol > TSEQ - 1) tcol = TSEQ - 1;          // masked anyway
            const unsigned short* kp = Kp + (size_t)tcol * HD;
#pragma unroll
            for (int c = 0; c < 2; ++c) {
                AFrag b;
                b.h[0] = *(const v8bf*)(kp + c * 32 + kbA);
                b.h[1] = *(const v8bf*)(kp + c * 32 + kbA + 16);
                s[half] = __builtin_amdgcn_wmma_f32_16x16x32_bf16(
                    false, qf[c].v, false, b.v, (short)0, s[half], false, false);
            }
        }
        // scale + causal mask + row max
        float rmax[8];
#pragma unroll
        for (int r = 0; r < 8; ++r) {
            const int qg = q0 + hb + r;
#pragma unroll
            for (int half = 0; half < 2; ++half) {
                int kg = kk + half * 16 + nl;
                float sv = s[half][r] * 0.125f;            // 1/sqrt(64)
                if (kg > qg) sv = -1e30f;
                s[half][r] = sv;
            }
            rmax[r] = fmaxf(s[0][r], s[1][r]);
        }
#pragma unroll
        for (int off = 8; off > 0; off >>= 1)
#pragma unroll
            for (int r = 0; r < 8; ++r)
                rmax[r] = fmaxf(rmax[r], __shfl_xor(rmax[r], off, 32));

        // online softmax update
        float alpha[8], rsum[8];
#pragma unroll
        for (int r = 0; r < 8; ++r) {
            float mn = fmaxf(mi[r], rmax[r]);
            alpha[r] = __expf(mi[r] - mn);
            mi[r]    = mn;
            float p0 = __expf(s[0][r] - mn);
            float p1 = __expf(s[1][r] - mn);
            s[0][r] = p0; s[1][r] = p1;
            rsum[r] = p0 + p1;
        }
#pragma unroll
        for (int off = 8; off > 0; off >>= 1)
#pragma unroll
            for (int r = 0; r < 8; ++r)
                rsum[r] += __shfl_xor(rsum[r], off, 32);
#pragma unroll
        for (int r = 0; r < 8; ++r) li[r] = li[r] * alpha[r] + rsum[r];
#pragma unroll
        for (int c4 = 0; c4 < 4; ++c4)
#pragma unroll
            for (int r = 0; r < 8; ++r) o[c4][r] *= alpha[r];

        // transpose P (C layout) -> A layout via LDS
        __syncthreads();
#pragma unroll
        for (int half = 0; half < 2; ++half)
#pragma unroll
            for (int r = 0; r < 8; ++r)
                pls[(hb + r) * 32 + half * 16 + nl] = f2bfu(s[half][r]);
        __syncthreads();
        AFrag pf;
        const unsigned short* pp = pls + nl * 32;
        pf.h[0] = *(const v8bf*)(pp + kbB);
        pf.h[1] = *(const v8bf*)(pp + kbB + 8);

        // O += P @ V  (V transposed [d][t] -> contiguous B-frags)
        int tb = kk + kbB;
        if (tb > TSEQ - 16) tb = TSEQ - 16;   // OOB rows multiply p==0
#pragma unroll
        for (int c4 = 0; c4 < 4; ++c4) {
            const int d = c4 * 16 + nl;
            const unsigned short* vp = Vp + (size_t)d * TSEQ + tb;
            AFrag b;
            b.h[0] = *(const v8bf*)(vp);
            b.h[1] = *(const v8bf*)(vp + 8);
            o[c4] = __builtin_amdgcn_wmma_f32_16x16x32_bf16(
                false, pf.v, false, b.v, (short)0, o[c4], false, false);
        }
    }

    // epilogue: normalize, store attended as bf16 [B,T,H*D]
    const int b_ = bh >> 4, h = bh & 15;
#pragma unroll
    for (int c4 = 0; c4 < 4; ++c4) {
        const int d = c4 * 16 + nl;
#pragma unroll
        for (int r = 0; r < 8; ++r) {
            int t = q0 + hb + r;
            float val = o[c4][r] / li[r];
            Ab[(size_t)(b_ * TSEQ + t) * CEMB + h * HD + d] = f2bfu(val);
        }
    }
}

// ---------------------------------------------------------------------------
extern "C" void kernel_launch(void* const* d_in, const int* in_sizes, int n_in,
                              void* d_out, int out_size, void* d_ws, size_t ws_size,
                              hipStream_t stream) {
    (void)in_sizes; (void)n_in; (void)out_size; (void)ws_size;

    const float* x  = (const float*)d_in[0];
    const float* Wq = (const float*)d_in[1];
    const float* bq = (const float*)d_in[2];
    const float* Wk = (const float*)d_in[3];
    const float* bk = (const float*)d_in[4];
    const float* Wv = (const float*)d_in[5];
    const float* bv = (const float*)d_in[6];
    const float* Wo = (const float*)d_in[7];
    const float* bo = (const float*)d_in[8];

    char* ws = (char*)d_ws;
    size_t off = 0;
    auto alloc = [&](size_t nelem) -> unsigned short* {
        unsigned short* p = (unsigned short*)(ws + off);
        off += nelem * sizeof(unsigned short);
        off = (off + 255) & ~(size_t)255;
        return p;
    };
    unsigned short* xb  = alloc((size_t)MROWS * CEMB);        // x in bf16
    unsigned short* wqb = alloc((size_t)CEMB * CEMB);
    unsigned short* wkb = alloc((size_t)CEMB * CEMB);
    unsigned short* wvb = alloc((size_t)CEMB * CEMB);
    unsigned short* wob = alloc((size_t)CEMB * CEMB);
    unsigned short* qb  = alloc((size_t)NB * NH * TSEQ * HD); // [B,H,T,D]
    unsigned short* kbf = alloc((size_t)NB * NH * TSEQ * HD); // [B,H,T,D]
    unsigned short* vtb = alloc((size_t)NB * NH * HD * TSEQ); // [B,H,D,T]
    unsigned short* ab  = alloc((size_t)MROWS * CEMB);        // attended bf16

    const int nx4 = MROWS * CEMB / 4;
    const int nw4 = CEMB * CEMB / 4;
    k_cvt4<<<(nx4 + 255) / 256, 256, 0, stream>>>((const float4*)x,  (unsigned long long*)xb,  nx4);
    k_cvt4<<<(nw4 + 255) / 256, 256, 0, stream>>>((const float4*)Wq, (unsigned long long*)wqb, nw4);
    k_cvt4<<<(nw4 + 255) / 256, 256, 0, stream>>>((const float4*)Wk, (unsigned long long*)wkb, nw4);
    k_cvt4<<<(nw4 + 255) / 256, 256, 0, stream>>>((const float4*)Wv, (unsigned long long*)wvb, nw4);
    k_cvt4<<<(nw4 + 255) / 256, 256, 0, stream>>>((const float4*)Wo, (unsigned long long*)wob, nw4);

    const int blocks = (MROWS / 128) * (CEMB / 64);  // 512 blocks, 8 waves each
    k_gemm<<<blocks, 256, 0, stream>>>(xb, wqb, bq, qb,  0);
    k_gemm<<<blocks, 256, 0, stream>>>(xb, wkb, bk, kbf, 0);
    k_gemm<<<blocks, 256, 0, stream>>>(xb, wvb, bv, vtb, 1);

    k_attn<<<NB * NH * (TSEQ / 16), 32, 0, stream>>>(qb, kbf, vtb, ab);

    k_gemm<<<blocks, 256, 0, stream>>>(ab, wob, bo, d_out, 2);
}
